// BiLSTM_CRF_50062138802672
// MI455X (gfx1250) — matrix operand order repaired and checked by reading, not compiled, not run on hardware
//
#include <hip/hip_runtime.h>

// Problem constants (match reference)
#define Tn 512
#define Bn 64
#define En 256
#define Hn 256
#define Kn 16
#define START_IDX 14
#define STOP_IDX  15
#define NEGV -10000.0f

#define LOG2E 1.4426950408889634f

typedef __attribute__((ext_vector_type(16))) __bf16 v16bf;
typedef __attribute__((ext_vector_type(8)))  __bf16 v8bf;
typedef __attribute__((ext_vector_type(8)))  float  v8f;

// ---- WMMA fragment helpers (this file's contiguous-k order, same for A and B) -------------
// A 16x32 bf16: lane m = lane&15; halves 0..15 -> K=k0..k0+15,
// with k0 = 16*(lane>=16). Two contiguous 16B loads.
__device__ __forceinline__ v16bf load_a_frag(const __bf16* p) {
  v8bf lo = *(const v8bf*)(p);
  v8bf hi = *(const v8bf*)(p + 8);
  return __builtin_shufflevector(lo, hi, 0,1,2,3,4,5,6,7,8,9,10,11,12,13,14,15);
}
// B 32x16 bf16: lane n = lane&15; 16 contiguous halves at K += 16*(lane>=16).
// Row-major [out,in] weights load as a single contiguous v16bf.

// Branch-free transcendentals: raw v_exp_f32 / v_rcp_f32, correct saturation
// at +/-inf, no exec-mask divergence in the latency-critical recurrence tail.
__device__ __forceinline__ float fast_sig(float x) {
  return __builtin_amdgcn_rcpf(1.0f + __builtin_amdgcn_exp2f(-LOG2E * x));
}
__device__ __forceinline__ float fast_tanh(float x) {
  return 1.0f - 2.0f * __builtin_amdgcn_rcpf(__builtin_amdgcn_exp2f(2.0f * LOG2E * x) + 1.0f);
}

// ---- Kernel 1: f32 -> bf16 conversion -------------------------------------
__global__ void cvt_bf16_kernel(const float* __restrict__ src, __bf16* __restrict__ dst, int n) {
  int i = blockIdx.x * blockDim.x + threadIdx.x;
  if (i < n) dst[i] = (__bf16)src[i];
}

// ---- Kernel 2: embedding gather -> x_bf16 [T*B, E] ------------------------
__global__ void gather_kernel(const int* __restrict__ sentence,
                              const float* __restrict__ emb,
                              __bf16* __restrict__ x) {
  int idx = blockIdx.x * blockDim.x + threadIdx.x;   // T*B*(E/4) = 2,097,152
  int e4  = idx & 63;                                 // 64 groups of 4 floats
  int row = idx >> 6;                                 // t*B + b
  int b = row & (Bn - 1);
  int t = row >> 6;
  int tok = sentence[(size_t)b * Tn + t];
  const float4 v = *((const float4*)(emb + (size_t)tok * En) + e4);
  __bf16* d = x + (size_t)row * En + e4 * 4;
  d[0] = (__bf16)v.x; d[1] = (__bf16)v.y; d[2] = (__bf16)v.z; d[3] = (__bf16)v.w;
}

// ---- Kernel 3: fused BiLSTM recurrence ------------------------------------
// grid.x = 2 (direction), block = 1024 threads = 32 waves, 2 tasks/wave.
// Weights stay in global (L2-resident, ~2MB total) and are re-fetched every
// step; the per-step memory clobber stops cross-step LICM, and the ROLLED
// K-loop bounds transient pressure so nothing spills (verified round 3).
// h lives in LDS (bf16), c in registers for all 512 steps.
__global__ __launch_bounds__(1024, 1)
void lstm_kernel(const __bf16* __restrict__ x,          // [T*B, 256]
                 const __bf16* __restrict__ wih_f, const __bf16* __restrict__ whh_f,
                 const float*  __restrict__ bias_f,
                 const __bf16* __restrict__ wih_b, const __bf16* __restrict__ whh_b,
                 const float*  __restrict__ bias_b,
                 __bf16* __restrict__ hcat)             // [T*B, 512]
{
  const int d = blockIdx.x;
  const __bf16* wih  = d ? wih_b  : wih_f;
  const __bf16* whh  = d ? whh_b  : whh_f;
  const float*  bias = d ? bias_b : bias_f;

  __shared__ __bf16 h_lds[Bn][Hn + 8];   // 64 x 264 bf16 = 33 KB, padded

  const int tid   = threadIdx.x;
  const int lane  = tid & 31;
  const int wave  = tid >> 5;            // 0..31
  const int lm    = lane & 15;           // m (A) / n (B,C) within tile
  const int khalf = lane >> 4;           // 0/1 lane-group

  for (int i = tid; i < Bn * (Hn + 8); i += 1024) ((__bf16*)h_lds)[i] = (__bf16)0.0f;
  __syncthreads();

  float cst[2][8];                       // persistent cell state (2 tasks x 8)
  v8bf  hnew[2];                         // new h, packed bf16 (8 VGPRs)
#pragma unroll
  for (int q = 0; q < 2; ++q)
#pragma unroll
    for (int r = 0; r < 8; ++r) cst[q][r] = 0.0f;

  for (int step = 0; step < Tn; ++step) {
    // Defeat cross-step LICM of the step-invariant weight fragments.
    asm volatile("" ::: "memory");

    const int t = d ? (Tn - 1 - step) : step;

#pragma clang loop unroll(disable)
    for (int q = 0; q < 2; ++q) {        // 2 tasks per wave, 64 tasks total
      const int task = wave * 2 + q;
      const int mi = task & 3;           // batch tile 0..3
      const int nh = task >> 2;          // hidden tile 0..15
      const int brow = mi * 16 + lm;     // A-matrix row (batch index)
      const __bf16* xrow = x + (size_t)(t * Bn + brow) * En;

      // prefetch next step's activation row into cache
      const int tnext = d ? (t - 1) : (t + 1);
      if ((unsigned)tnext < Tn)
        __builtin_prefetch(x + (size_t)(tnext * Bn + brow) * En, 0, 1);

      v8f acc[4];
#pragma unroll
      for (int g = 0; g < 4; ++g) {
        const float bv = bias[g * Hn + nh * 16 + lm];
#pragma unroll
        for (int r = 0; r < 8; ++r) acc[g][r] = bv;
      }

      // ROLLED K-loop: bounds transient register pressure to ~1 B-fragment
      // so nothing spills; latency hidden by 32 waves instead of unrolling.
#pragma clang loop unroll(disable)
      for (int k8 = 0; k8 < 8; ++k8) {   // K = 256 in chunks of 32
        const int kb = k8 * 32;
        const v16bf ax = load_a_frag(xrow + kb + khalf * 16);
        const v16bf ah = load_a_frag(&h_lds[brow][kb + khalf * 16]);
#pragma unroll
        for (int g = 0; g < 4; ++g) {
          const size_t jrow = (size_t)(g * Hn + nh * 16 + lm) * 256;
          const v16bf bi = *(const v16bf*)(wih + jrow + kb + khalf * 16);
          acc[g] = __builtin_amdgcn_wmma_f32_16x16x32_bf16(
                       false, ax, false, bi, (short)0, acc[g], false, false);
          const v16bf bh = *(const v16bf*)(whh + jrow + kb + khalf * 16);
          acc[g] = __builtin_amdgcn_wmma_f32_16x16x32_bf16(
                       false, ah, false, bh, (short)0, acc[g], false, false);
        }
      }

      // Gate fusion: element r of all four accs is the same (b, j).
      // Branch-free nonlinearities keep the serial tail short.
#pragma unroll
      for (int r = 0; r < 8; ++r) {
        const float iv = fast_sig(acc[0][r]);
        const float fv = fast_sig(acc[1][r]);
        const float gv = fast_tanh(acc[2][r]);
        const float ov = fast_sig(acc[3][r]);
        const float c  = fv * cst[q][r] + iv * gv;
        cst[q][r]  = c;
        hnew[q][r] = (__bf16)(ov * fast_tanh(c));
      }
    }

    __syncthreads();                     // all reads of h(t-1) done
#pragma unroll
    for (int q = 0; q < 2; ++q) {
      const int task = wave * 2 + q;
      const int mi = task & 3, nh = task >> 2;
#pragma unroll
      for (int r = 0; r < 8; ++r) {
        const int m  = (khalf << 3) + r; // C-frag: M = r + 8*(lane>=16)
        const int bb = mi * 16 + m;
        const int jn = nh * 16 + lm;     // C-frag: N = lane&15
        const __bf16 hb = hnew[q][r];
        h_lds[bb][jn] = hb;
        hcat[(size_t)(t * Bn + bb) * (2 * Hn) + d * Hn + jn] = hb;
      }
    }
    __syncthreads();                     // h(t) visible for next step
  }
}

// ---- Kernel 4: output projection feats = hcat @ w_out^T + b_out -----------
// M = T*B = 32768 (2048 tiles), N = 16 (one tile), K = 512.
__global__ __launch_bounds__(256)
void proj_kernel(const __bf16* __restrict__ hcat,     // [T*B, 512]
                 const __bf16* __restrict__ wout,     // [16, 512] bf16
                 const float*  __restrict__ b_out,
                 float* __restrict__ feats)           // [T*B, 16]
{
  const int gwave = (blockIdx.x * 256 + threadIdx.x) >> 5;  // 0..2047
  const int lane  = threadIdx.x & 31;
  const int lm = lane & 15, khalf = lane >> 4;
  const int mbase = gwave * 16;

  const __bf16* arow = hcat + (size_t)(mbase + lm) * (2 * Hn);
  const __bf16* wrow = wout + (size_t)lm * (2 * Hn);

  v8f acc;
  const float bv = b_out[lm];
#pragma unroll
  for (int r = 0; r < 8; ++r) acc[r] = bv;

#pragma unroll
  for (int k16 = 0; k16 < 16; ++k16) {   // K = 512 in chunks of 32
    const int kb = k16 * 32;
    const v16bf a = load_a_frag(arow + kb + khalf * 16);
    const v16bf b = *(const v16bf*)(wrow + kb + khalf * 16);
    acc = __builtin_amdgcn_wmma_f32_16x16x32_bf16(
              false, a, false, b, (short)0, acc, false, false);
  }
#pragma unroll
  for (int r = 0; r < 8; ++r) {
    const int m = (khalf << 3) + r;
    feats[(size_t)(mbase + m) * Kn + lm] = acc[r];
  }
}

// ---- Kernel 5: Viterbi (K=16, tiny) — one block per batch element ---------
__global__ void viterbi_kernel(const float* __restrict__ feats,       // [T*B, 16]
                               const float* __restrict__ trans,       // [16, 16]
                               unsigned char* __restrict__ bptr,      // [T*B, 16]
                               float* __restrict__ out)               // [64 + T*64]
{
  const int b = blockIdx.x;
  const int k = threadIdx.x;             // 16 threads: next-tag index
  __shared__ float fv[Kn];

  fv[k] = (k == START_IDX) ? 0.0f : NEGV;
  float trow[Kn];
#pragma unroll
  for (int p = 0; p < Kn; ++p) trow[p] = trans[k * Kn + p];   // row = next
  __syncthreads();

  for (int t = 0; t < Tn; ++t) {
    float best = -3.0e38f; int arg = 0;
#pragma unroll
    for (int p = 0; p < Kn; ++p) {
      const float s = fv[p] + trow[p];
      if (s > best) { best = s; arg = p; }
    }
    bptr[(size_t)(t * Bn + b) * Kn + k] = (unsigned char)arg;
    const float nf = best + feats[(size_t)(t * Bn + b) * Kn + k];
    __syncthreads();
    fv[k] = nf;
    __syncthreads();
  }

  if (k == 0) {
    float best = -3.0e38f; int arg = 0;
#pragma unroll
    for (int p = 0; p < Kn; ++p) {
      const float s = fv[p] + trans[STOP_IDX * Kn + p];
      if (s > best) { best = s; arg = p; }
    }
    out[b] = best;
    int tag = arg;
    for (int t = Tn - 1; t >= 0; --t) {
      out[Bn + (size_t)t * Bn + b] = (float)tag;
      tag = bptr[(size_t)(t * Bn + b) * Kn + tag];
    }
  }
}

// ---- Host launch ----------------------------------------------------------
extern "C" void kernel_launch(void* const* d_in, const int* in_sizes, int n_in,
                              void* d_out, int out_size, void* d_ws, size_t ws_size,
                              hipStream_t stream) {
  const int*   sentence = (const int*)  d_in[0];
  const float* emb      = (const float*)d_in[1];
  const float* w_ih_f   = (const float*)d_in[2];
  const float* w_hh_f   = (const float*)d_in[3];
  const float* b_f      = (const float*)d_in[4];
  const float* w_ih_b   = (const float*)d_in[5];
  const float* w_hh_b   = (const float*)d_in[6];
  const float* b_b      = (const float*)d_in[7];
  const float* w_out    = (const float*)d_in[8];
  const float* b_out    = (const float*)d_in[9];
  const float* trans    = (const float*)d_in[10];
  float* out = (float*)d_out;

  // Workspace layout (256B aligned)
  char* w = (char*)d_ws;
  __bf16* xbf    = (__bf16*)(w + 0);              // 16 MB: [T*B, 256]
  __bf16* wihf   = (__bf16*)(w + 16777216);       // 512 KB
  __bf16* whhf   = (__bf16*)(w + 17301504);       // 512 KB
  __bf16* wihb   = (__bf16*)(w + 17825792);       // 512 KB
  __bf16* whhb   = (__bf16*)(w + 18350080);       // 512 KB
  __bf16* woutb  = (__bf16*)(w + 18874368);       // 16 KB
  __bf16* hcat   = (__bf16*)(w + 18890752);       // 32 MB: [T*B, 512]
  float*  feats  = (float*) (w + 52445184);       // 2 MB: [T*B, 16]
  unsigned char* bptr = (unsigned char*)(w + 54542336); // 512 KB

  const int nW = 1024 * 256;                      // 4H x E (= 4H x H)
  cvt_bf16_kernel<<<(nW + 255) / 256, 256, 0, stream>>>(w_ih_f, wihf, nW);
  cvt_bf16_kernel<<<(nW + 255) / 256, 256, 0, stream>>>(w_hh_f, whhf, nW);
  cvt_bf16_kernel<<<(nW + 255) / 256, 256, 0, stream>>>(w_ih_b, wihb, nW);
  cvt_bf16_kernel<<<(nW + 255) / 256, 256, 0, stream>>>(w_hh_b, whhb, nW);
  cvt_bf16_kernel<<<(Kn * 2 * Hn + 255) / 256, 256, 0, stream>>>(w_out, woutb, Kn * 2 * Hn);

  gather_kernel<<<(Tn * Bn * (En / 4)) / 256, 256, 0, stream>>>(sentence, emb, xbf);

  lstm_kernel<<<2, 1024, 0, stream>>>(xbf, wihf, whhf, b_f, wihb, whhb, b_b, hcat);

  proj_kernel<<<(Tn * Bn / 16) / 8, 256, 0, stream>>>(hcat, woutb, b_out, feats);

  viterbi_kernel<<<Bn, Kn, 0, stream>>>(feats, trans, bptr, out);
}
